// SpikformerPrune_38173669327039
// MI455X (gfx1250) — matrix operand order repaired and checked by reading, not compiled, not run on hardware
//
#include <hip/hip_runtime.h>

// =====================================================================
// Spikformer-with-token-pruning forward for MI455X (gfx1250, wave32).
// GEMM-like ops use v_wmma_f32_16x16x32_f16. Activation tensors that
// feed WMMA as the B operand are stored token-major (img, N=64, C) so
// every lane's B fragment is one contiguous 32B run (global_load_b128);
// weights stay row-major (M,K) so A fragments are contiguous too.
// LIF (tau=2, hard reset) fused into BN epilogues; SPS convs are VALU
// on u8 spikes.
// =====================================================================

typedef __attribute__((ext_vector_type(16))) _Float16 v16h;
typedef __attribute__((ext_vector_type(8)))  _Float16 v8h;
typedef __attribute__((ext_vector_type(8)))  float    v8f;
typedef _Float16 half_t;
typedef unsigned char u8;

#define BN_EPS 1e-5f

static __device__ __forceinline__ v16h cat16(v8h lo, v8h hi){
  return __builtin_shufflevector(lo, hi, 0,1,2,3,4,5,6,7,8,9,10,11,12,13,14,15);
}
static __device__ __forceinline__ v8h ld8(const half_t* p){
  return *(const v8h*)p;   // 16B aligned by construction
}

// ---------------- f32 -> f16 weight conversion ----------------
__global__ void k_cvt_f16(const float* __restrict__ s, half_t* __restrict__ d, int n){
  int i = blockIdx.x * blockDim.x + threadIdx.x;
  if (i < n) d[i] = (half_t)s[i];
}

// ---------------- SPS stage 0: conv3x3(2->48)+BN+LIF+maxpool3x3s2 fused ----------------
__global__ void k_conv0_pool(const float* __restrict__ x, const float* __restrict__ w,
                             const float* __restrict__ gg, const float* __restrict__ bb,
                             const float* __restrict__ mm, const float* __restrict__ vv,
                             u8* __restrict__ out){
  int idx = blockIdx.x * blockDim.x + threadIdx.x;
  const int HWo = 64 * 64;
  if (idx >= 16 * 48 * HWo) return;
  int pw = idx & 63;
  int ph = (idx >> 6) & 63;
  int co = (idx / HWo) % 48;
  int b  = idx / (48 * HWo);
  float scale = gg[co] * rsqrtf(vv[co] + BN_EPS);
  float mean = mm[co], beta = bb[co];
  u8 sp0 = 0, sp1 = 0, sp2 = 0, sp3 = 0;
  for (int dy = -1; dy <= 1; ++dy){
    int h = 2 * ph + dy;
    if (h < 0 || h >= 128) continue;
    for (int dx = -1; dx <= 1; ++dx){
      int wv = 2 * pw + dx;
      if (wv < 0 || wv >= 128) continue;
      float vmem = 0.f;
      for (int t = 0; t < 4; ++t){
        float acc = 0.f;
        for (int ci = 0; ci < 2; ++ci){
          const float* xc = x + ((size_t)((t * 16 + b) * 2 + ci)) * 16384;
          const float* wc = w + (co * 2 + ci) * 9;
          for (int kh = 0; kh < 3; ++kh){
            int hh = h + kh - 1;
            if (hh < 0 || hh >= 128) continue;
            for (int kw = 0; kw < 3; ++kw){
              int ww = wv + kw - 1;
              if (ww < 0 || ww >= 128) continue;
              acc += wc[kh * 3 + kw] * xc[hh * 128 + ww];
            }
          }
        }
        float xb = (acc - mean) * scale + beta;
        float hm = vmem + (xb - vmem) * 0.5f;
        float s = (hm >= 1.f) ? 1.f : 0.f;
        vmem = hm * (1.f - s);
        u8 su = (u8)s;
        if (t == 0) sp0 |= su; else if (t == 1) sp1 |= su;
        else if (t == 2) sp2 |= su; else sp3 |= su;
      }
    }
  }
  size_t o  = (size_t)co * HWo + ph * 64 + pw;
  size_t cs = (size_t)48 * HWo;
  out[((size_t)(0 * 16 + b)) * cs + o] = sp0;
  out[((size_t)(1 * 16 + b)) * cs + o] = sp1;
  out[((size_t)(2 * 16 + b)) * cs + o] = sp2;
  out[((size_t)(3 * 16 + b)) * cs + o] = sp3;
}

// ---------------- generic conv3x3 SAME on u8 spikes + BN + LIF -> u8 spikes ----------------
__global__ void k_conv_bn_lif(const u8* __restrict__ in, const float* __restrict__ w,
                              const float* __restrict__ gg, const float* __restrict__ bb,
                              const float* __restrict__ mm, const float* __restrict__ vv,
                              u8* __restrict__ out, int Ci, int Co, int H){
  int idx = blockIdx.x * blockDim.x + threadIdx.x;
  int HW = H * H;
  if (idx >= 16 * Co * HW) return;
  int wv = idx % H;
  int h  = (idx / H) % H;
  int co = (idx / HW) % Co;
  int b  = idx / (Co * HW);
  float scale = gg[co] * rsqrtf(vv[co] + BN_EPS);
  float mean = mm[co], beta = bb[co];
  const float* wc0 = w + (size_t)co * Ci * 9;
  float vmem = 0.f;
  for (int t = 0; t < 4; ++t){
    const u8* ib = in + ((size_t)(t * 16 + b)) * Ci * HW;
    float acc = 0.f;
    for (int ci = 0; ci < Ci; ++ci){
      const u8* ic = ib + (size_t)ci * HW;
      const float* wc = wc0 + ci * 9;
      for (int kh = 0; kh < 3; ++kh){
        int hh = h + kh - 1;
        if (hh < 0 || hh >= H) continue;
        const u8* ir = ic + hh * H;
        for (int kw = 0; kw < 3; ++kw){
          int ww = wv + kw - 1;
          if (ww < 0 || ww >= H) continue;
          if (ir[ww]) acc += wc[kh * 3 + kw];
        }
      }
    }
    float xb = (acc - mean) * scale + beta;
    float hm = vmem + (xb - vmem) * 0.5f;
    float s = (hm >= 1.f) ? 1.f : 0.f;
    vmem = hm * (1.f - s);
    out[((size_t)(t * 16 + b) * Co + co) * HW + h * H + wv] = (u8)s;
  }
}

// ---------------- maxpool 3x3 s2 p1 on u8 spikes ----------------
__global__ void k_pool_u8(const u8* __restrict__ in, u8* __restrict__ out, int C, int H){
  int Ho = H >> 1;
  int idx = blockIdx.x * blockDim.x + threadIdx.x;
  if (idx >= 64 * C * Ho * Ho) return;
  int pw  = idx % Ho;
  int ph  = (idx / Ho) % Ho;
  int c   = (idx / (Ho * Ho)) % C;
  int img = idx / (C * Ho * Ho);
  const u8* ib = in + ((size_t)img * C + c) * H * H;
  u8 m = 0;
  for (int dy = -1; dy <= 1; ++dy){
    int h = 2 * ph + dy;
    if (h < 0 || h >= H) continue;
    for (int dx = -1; dx <= 1; ++dx){
      int wv = 2 * pw + dx;
      if (wv < 0 || wv >= H) continue;
      m |= ib[h * H + wv];
    }
  }
  out[idx] = m;
}

// ---------------- feats = z + rpe, transpose (img,c,n)->(img,n,c), u8->f16 ----------------
__global__ void k_make_feats(const u8* __restrict__ z, const u8* __restrict__ rp,
                             half_t* __restrict__ feats){
  int idx = blockIdx.x * blockDim.x + threadIdx.x;
  if (idx >= 64 * 384 * 64) return;
  int n = idx & 63;
  int c = (idx >> 6) % 384;
  int img = idx / 24576;
  feats[(size_t)img * 24576 + n * 384 + c] = (half_t)((float)(z[idx] + rp[idx]));
}

// ---------------- batched GEMM: OUT[g](MxN=64,f32) = A(MxK,f16) x B[g] ----------------
// B stored token-major (64 x K) => B fragment is contiguous per lane.
// One wave = one m-tile x two n-tiles (A reuse, 2 wmma per K step).
__global__ void k_gemm_wx(const half_t* __restrict__ A, const half_t* __restrict__ B,
                          float* __restrict__ OUT, int M, int K, int tilesM){
  int wid  = (blockIdx.x * blockDim.x + threadIdx.x) >> 5;
  int lane = threadIdx.x & 31;
  int tpg = tilesM * 2;
  int g = wid / tpg;
  int r = wid - g * tpg;
  int m0 = (r >> 1) << 4;
  int n0 = (r & 1) << 5;                 // 0 or 32
  const half_t* Bg = B + (size_t)g * 64 * K;
  int ml  = lane & 15;
  int kbA = (lane >> 4) << 4;            // 0 / 16
  int kbB = (lane >> 4) << 4;            // 0 / 16
  v8f acc0 = {}, acc1 = {};
  for (int k0 = 0; k0 < K; k0 += 32){
    const half_t* Ap  = A  + (size_t)(m0 + ml) * K + k0 + kbA;
    const half_t* Bp0 = Bg + (size_t)(n0 + ml) * K + k0 + kbB;
    const half_t* Bp1 = Bp0 + (size_t)16 * K;
    __builtin_prefetch(Ap + 32, 0, 1);
    v16h a  = cat16(ld8(Ap),      ld8(Ap + 8));
    v16h b0 = cat16(ld8(Bp0),     ld8(Bp0 + 8));
    v16h b1 = cat16(ld8(Bp1),     ld8(Bp1 + 8));
    acc0 = __builtin_amdgcn_wmma_f32_16x16x32_f16(false, a, false, b0, (short)0, acc0, false, false);
    acc1 = __builtin_amdgcn_wmma_f32_16x16x32_f16(false, a, false, b1, (short)0, acc1, false, false);
  }
  float* Og = OUT + (size_t)g * M * 64;
  int dm = (lane >> 4) << 3;
#pragma unroll
  for (int i = 0; i < 8; ++i){
    Og[(m0 + dm + i) * 64 + n0 + ml]      = acc0[i];
    Og[(m0 + dm + i) * 64 + n0 + 16 + ml] = acc1[i];
  }
}

// ---------------- attn[n,m] = sum_d q[n,d] k[m,d]; q,k token-major; K=32 ----------------
__global__ void k_attn_qk(const half_t* __restrict__ QS, const half_t* __restrict__ KS,
                          const float* __restrict__ mask, half_t* __restrict__ ATTN){
  int wid  = (blockIdx.x * blockDim.x + threadIdx.x) >> 5;
  int lane = threadIdx.x & 31;
  int g2 = wid >> 4;                     // (tb, head)
  int r  = wid & 15;                     // 4x4 tiles of 64x64
  int n0 = (r >> 2) << 4;
  int m0 = (r & 3) << 4;
  int tb = g2 / 12;
  int hh = g2 - tb * 12;
  int bidx = tb & 15;
  int ml  = lane & 15;
  int kbA = (lane >> 4) << 4;
  int kbB = (lane >> 4) << 4;
  const half_t* Qrow = QS + (size_t)tb * 24576 + (size_t)(n0 + ml) * 384 + hh * 32;
  const half_t* Krow = KS + (size_t)tb * 24576 + (size_t)(m0 + ml) * 384 + hh * 32;
  v16h a = cat16(ld8(Qrow + kbA), ld8(Qrow + 8 + kbA));
  v16h b = cat16(ld8(Krow + kbB), ld8(Krow + kbB + 8));
  v8f acc = {};
  acc = __builtin_amdgcn_wmma_f32_16x16x32_f16(false, a, false, b, (short)0, acc, false, false);
  half_t* Og = ATTN + (size_t)g2 * 4096;
  int dm = (lane >> 4) << 3;
#pragma unroll
  for (int i = 0; i < 8; ++i){
    int nr = n0 + dm + i;
    int mc = m0 + ml;
    float vx = acc[i];
    if (mask){
      if (!((mask[bidx * 64 + mc] != 0.f) || (nr == mc))) vx = 0.f;
    }
    Og[nr * 64 + mc] = (half_t)vx;
  }
}

// ---------------- y[c=h*32+d, n] = 0.25*sum_m attn[n,m] v[c,m]; v channel-major; K=64 ----------------
__global__ void k_attn_av(const half_t* __restrict__ ATTN, const half_t* __restrict__ VS,
                          float* __restrict__ YPRE){
  int wid  = (blockIdx.x * blockDim.x + threadIdx.x) >> 5;
  int lane = threadIdx.x & 31;
  int g2 = wid >> 3;
  int r  = wid & 7;
  int n0 = (r >> 1) << 4;                // token rows
  int d0 = (r & 1) << 4;                 // head-dim cols
  int tb = g2 / 12;
  int hh = g2 - tb * 12;
  int ml  = lane & 15;
  int kbA = (lane >> 4) << 4;
  int kbB = (lane >> 4) << 4;
  const half_t* Arow = ATTN + (size_t)g2 * 4096 + (size_t)(n0 + ml) * 64;
  const half_t* Vrow = VS + (size_t)tb * 24576 + (size_t)(hh * 32 + d0 + ml) * 64;
  v8f acc = {};
  for (int k0 = 0; k0 < 64; k0 += 32){
    v16h a = cat16(ld8(Arow + k0 + kbA), ld8(Arow + k0 + kbA + 8));
    v16h b = cat16(ld8(Vrow + k0 + kbB), ld8(Vrow + k0 + kbB + 8));
    acc = __builtin_amdgcn_wmma_f32_16x16x32_f16(false, a, false, b, (short)0, acc, false, false);
  }
  int dm = (lane >> 4) << 3;
  float* Og = YPRE + (size_t)tb * 24576 + (size_t)hh * 2048;
#pragma unroll
  for (int i = 0; i < 8; ++i){
    int nr = n0 + dm + i;
    int dc = d0 + ml;
    Og[dc * 64 + nr] = 0.25f * acc[i];   // (c, n) layout for the LIF transpose epilogue
  }
}

// ---------------- (bias) + (BN) + LIF over T + (residual); optional transpose to token-major ----------------
// pre: f32 (t*16+b, C, 64). tout=1: out (t*16+b, 64, C); tout=0: out (t*16+b, C, 64).
__global__ void k_bn_lif(const float* __restrict__ pre,
                         const float* __restrict__ gg, const float* __restrict__ bb,
                         const float* __restrict__ mm, const float* __restrict__ vv,
                         const float* __restrict__ bias, float vth,
                         const half_t* __restrict__ res, half_t* __restrict__ out,
                         int C, int tout){
  int idx = blockIdx.x * blockDim.x + threadIdx.x;
  int S = 16 * C * 64;
  if (idx >= S) return;
  int n = idx & 63;
  int c = (idx >> 6) % C;
  int b = idx / (C * 64);
  size_t obase = tout ? ((size_t)b * C * 64 + (size_t)n * C + c) : (size_t)idx;
  float scale = 1.f, mean = 0.f, beta = 0.f, bs = 0.f;
  if (gg){ scale = gg[c] * rsqrtf(vv[c] + BN_EPS); mean = mm[c]; beta = bb[c]; }
  if (bias) bs = bias[c];
  float vmem = 0.f;
  for (int t = 0; t < 4; ++t){
    float xv = pre[(size_t)t * S + idx] + bs;
    if (gg) xv = (xv - mean) * scale + beta;
    float hm = vmem + (xv - vmem) * 0.5f;
    float s = (hm >= vth) ? 1.f : 0.f;
    vmem = hm * (1.f - s);
    float o = s;
    if (res) o += (float)res[(size_t)t * S + obase];
    out[(size_t)t * S + obase] = (half_t)o;
  }
}

// ---------------- predictor: time mean (token-major), fc1+gelu, score+top-44 ----------------
__global__ void k_time_mean(const half_t* __restrict__ feats, float* __restrict__ mean){
  int idx = blockIdx.x * blockDim.x + threadIdx.x;
  const int S = 16 * 384 * 64;
  if (idx >= S) return;
  float acc = 0.f;
  for (int t = 0; t < 4; ++t) acc += (float)feats[(size_t)t * S + idx];
  mean[idx] = 0.25f * acc;               // keeps (b, n, c) layout
}

__global__ void k_pred_fc1(const float* __restrict__ mean, const float* __restrict__ w,
                           const float* __restrict__ bv, float* __restrict__ H1){
  int bn = blockIdx.x;                   // b*64+n
  int j = threadIdx.x;                   // 0..191
  const float* mb = mean + (size_t)bn * 384;   // contiguous channel row
  const float* wr = w + (size_t)j * 384;
  float s = bv[j];
  for (int c = 0; c < 384; ++c) s += mb[c] * wr[c];
  H1[(size_t)bn * 192 + j] = 0.5f * s * (1.f + erff(s * 0.70710678f));
}

__global__ void k_score_topk(const float* __restrict__ H1, const float* __restrict__ l2w,
                             const float* __restrict__ l2b, float* __restrict__ mask){
  __shared__ float sd[64];
  int b = blockIdx.x, n = threadIdx.x;
  const float* hb = H1 + ((size_t)b * 64 + n) * 192;
  float l0 = l2b[0], l1 = l2b[1];
  for (int j = 0; j < 192; ++j){ float hv = hb[j]; l0 += hv * l2w[j]; l1 += hv * l2w[192 + j]; }
  sd[n] = l0 - l1;
  __syncthreads();
  float mine = sd[n];
  int rank = 0;
  for (int m = 0; m < 64; ++m){
    float o = sd[m];
    rank += (o > mine) || (o == mine && m < n);
  }
  mask[b * 64 + n] = (rank < 44) ? 1.f : 0.f;
}

// ---------------- masked token mean + classifier head ----------------
__global__ void k_pool_tokens(const half_t* __restrict__ feats, const float* __restrict__ mask,
                              float* __restrict__ pooled){
  int idx = blockIdx.x * blockDim.x + threadIdx.x;
  if (idx >= 16 * 384) return;
  int c = idx % 384, b = idx / 384;
  const float* mkb = mask + b * 64;
  float acc = 0.f;
  for (int t = 0; t < 4; ++t){
    const half_t* fb = feats + (size_t)(t * 16 + b) * 24576;
    for (int n = 0; n < 64; ++n) acc += mkb[n] * (float)fb[n * 384 + c];
  }
  pooled[idx] = acc * (1.f / (4.f * 44.f));
}

__global__ void k_head(const float* __restrict__ pooled, const float* __restrict__ w,
                       const float* __restrict__ bv, float* __restrict__ out){
  int idx = blockIdx.x * blockDim.x + threadIdx.x;
  if (idx >= 16 * 11) return;
  int k = idx % 11, b = idx / 11;
  float acc = bv[k];
  const float* pb = pooled + b * 384;
  const float* wr = w + k * 384;
  for (int c = 0; c < 384; ++c) acc += pb[c] * wr[c];
  out[idx] = acc;
}

// =====================================================================
// Host side
// =====================================================================
struct BNp { const float *g, *b, *m, *v; };

extern "C" void kernel_launch(void* const* d_in, const int* in_sizes, int n_in,
                              void* d_out, int out_size, void* d_ws, size_t ws_size,
                              hipStream_t stream){
  (void)in_sizes; (void)n_in; (void)out_size; (void)ws_size;
  // ---- input mapping: setup_inputs() dict insertion order, recursively flattened ----
  int p = 0;
  auto nf  = [&]() -> const float* { return (const float*)d_in[p++]; };
  auto nbn = [&]() -> BNp { BNp r; r.g = nf(); r.b = nf(); r.m = nf(); r.v = nf(); return r; };
  const float* x    = nf();
  const float* c0w  = nf(); BNp bn0 = nbn();
  const float* c1w  = nf(); BNp bn1 = nbn();
  const float* c2w  = nf(); BNp bn2 = nbn();
  const float* c3w  = nf(); BNp bn3 = nbn();
  const float* rpew = nf(); BNp rpebn = nbn();
  const float *qw[4], *kw[4], *vw[4], *pjw[4], *pjb[4], *f1w[4], *f1b[4], *f2w[4], *f2b[4];
  BNp qbn[4], kbn[4], vbn[4], pjbn[4], f1bn[4], f2bn[4];
  for (int i = 0; i < 4; ++i){
    qw[i]  = nf(); qbn[i]  = nbn();
    kw[i]  = nf(); kbn[i]  = nbn();
    vw[i]  = nf(); vbn[i]  = nbn();
    pjw[i] = nf(); pjb[i]  = nf(); pjbn[i] = nbn();
    f1w[i] = nf(); f1b[i]  = nf(); f1bn[i] = nbn();
    f2w[i] = nf(); f2b[i]  = nf(); f2bn[i] = nbn();
  }
  const float* l1w = nf(); const float* l1b = nf();
  const float* l2w = nf(); const float* l2b = nf();
  const float* hww = nf(); const float* hbb = nf();

  // ---- workspace layout (~91 MB total, explicit reuse) ----
  char* wsb = (char*)d_ws;
  const size_t MB = 1024 * 1024;
  half_t* W16    = (half_t*)(wsb);              // f16 weights: 14.2MB
  half_t* FEATS  = (half_t*)(wsb + 16 * MB);    // (img,64,384) f16
  half_t* QS     = (half_t*)(wsb + 20 * MB);    // q spikes / y spikes (token-major)
  half_t* KS     = (half_t*)(wsb + 24 * MB);
  half_t* VS     = (half_t*)(wsb + 28 * MB);    // v spikes (channel-major)
  half_t* ATTN   = (half_t*)(wsb + 32 * MB);
  float*  PRE    = (float*)(wsb + 40 * MB);
  u8*     R1     = (u8*)(wsb + 48 * MB);        // 26MB: conv outs / fc1 pre f32
  u8*     R2     = (u8*)(wsb + 74 * MB);        // 13MB: pooled spikes / mlp hidden spikes
  float*  MEAN   = (float*)(wsb + 87 * MB);
  float*  H1B    = (float*)(wsb + 89 * MB);
  float*  MASK   = (float*)(wsb + 90 * MB);
  float*  POOLED = (float*)(wsb + 90 * MB + 65536);

  // ---- convert block GEMM weights to f16 (row-major M x K kept) ----
  const size_t BW = 1769472;
  for (int i = 0; i < 4; ++i){
    half_t* bwp = W16 + (size_t)i * BW;
    k_cvt_f16<<<(147456 + 255) / 256, 256, 0, stream>>>(qw[i],  bwp,            147456);
    k_cvt_f16<<<(147456 + 255) / 256, 256, 0, stream>>>(kw[i],  bwp + 147456,   147456);
    k_cvt_f16<<<(147456 + 255) / 256, 256, 0, stream>>>(vw[i],  bwp + 294912,   147456);
    k_cvt_f16<<<(147456 + 255) / 256, 256, 0, stream>>>(pjw[i], bwp + 442368,   147456);
    k_cvt_f16<<<(589824 + 255) / 256, 256, 0, stream>>>(f1w[i], bwp + 589824,   589824);
    k_cvt_f16<<<(589824 + 255) / 256, 256, 0, stream>>>(f2w[i], bwp + 1179648,  589824);
  }

  // ---- SPS stem ----
  k_conv0_pool<<<12288, 256, 0, stream>>>(x, c0w, bn0.g, bn0.b, bn0.m, bn0.v, R2);
  k_conv_bn_lif<<<24576, 256, 0, stream>>>(R2, c1w, bn1.g, bn1.b, bn1.m, bn1.v, R1, 48, 96, 64);
  k_pool_u8<<<24576, 256, 0, stream>>>(R1, R2, 96, 64);
  k_conv_bn_lif<<<12288, 256, 0, stream>>>(R2, c2w, bn2.g, bn2.b, bn2.m, bn2.v, R1, 96, 192, 32);
  k_pool_u8<<<12288, 256, 0, stream>>>(R1, R2, 192, 32);
  k_conv_bn_lif<<<6144, 256, 0, stream>>>(R2, c3w, bn3.g, bn3.b, bn3.m, bn3.v, R1, 192, 384, 16);
  k_pool_u8<<<6144, 256, 0, stream>>>(R1, R2, 384, 16);
  k_conv_bn_lif<<<1536, 256, 0, stream>>>(R2, rpew, rpebn.g, rpebn.b, rpebn.m, rpebn.v, R1, 384, 384, 8);
  k_make_feats<<<6144, 256, 0, stream>>>(R2, R1, FEATS);

  // ---- transformer blocks (prune at block 1) ----
  for (int i = 0; i < 4; ++i){
    if (i == 1){
      k_time_mean<<<1536, 256, 0, stream>>>(FEATS, MEAN);
      k_pred_fc1<<<1024, 192, 0, stream>>>(MEAN, l1w, l1b, H1B);
      k_score_topk<<<16, 64, 0, stream>>>(H1B, l2w, l2b, MASK);
    }
    const float* maskp = (i >= 1) ? MASK : nullptr;
    half_t* bwp = W16 + (size_t)i * BW;

    // q, k, v: pconv + BN + LIF   (grid: tilesM*2*64 waves / 8)
    k_gemm_wx<<<384, 256, 0, stream>>>(bwp,          FEATS, PRE, 384, 384, 24);
    k_bn_lif<<<1536, 256, 0, stream>>>(PRE, qbn[i].g, qbn[i].b, qbn[i].m, qbn[i].v, nullptr, 1.f, nullptr, QS, 384, 1);
    k_gemm_wx<<<384, 256, 0, stream>>>(bwp + 147456, FEATS, PRE, 384, 384, 24);
    k_bn_lif<<<1536, 256, 0, stream>>>(PRE, kbn[i].g, kbn[i].b, kbn[i].m, kbn[i].v, nullptr, 1.f, nullptr, KS, 384, 1);
    k_gemm_wx<<<384, 256, 0, stream>>>(bwp + 294912, FEATS, PRE, 384, 384, 24);
    k_bn_lif<<<1536, 256, 0, stream>>>(PRE, vbn[i].g, vbn[i].b, vbn[i].m, vbn[i].v, nullptr, 1.f, nullptr, VS, 384, 0);

    // spiking attention
    k_attn_qk<<<1536, 256, 0, stream>>>(QS, KS, maskp, ATTN);
    k_attn_av<<<768, 256, 0, stream>>>(ATTN, VS, PRE);
    k_bn_lif<<<1536, 256, 0, stream>>>(PRE, nullptr, nullptr, nullptr, nullptr, nullptr, 0.5f, nullptr, QS, 384, 1);
    k_gemm_wx<<<384, 256, 0, stream>>>(bwp + 442368, QS, PRE, 384, 384, 24);
    k_bn_lif<<<1536, 256, 0, stream>>>(PRE, pjbn[i].g, pjbn[i].b, pjbn[i].m, pjbn[i].v, pjb[i], 1.f, FEATS, FEATS, 384, 1);

    // MLP
    float*  FC1PRE = (float*)R1;
    half_t* HSPK   = (half_t*)R2;
    k_gemm_wx<<<1536, 256, 0, stream>>>(bwp + 589824, FEATS, FC1PRE, 1536, 384, 96);
    k_bn_lif<<<6144, 256, 0, stream>>>(FC1PRE, f1bn[i].g, f1bn[i].b, f1bn[i].m, f1bn[i].v, f1b[i], 1.f, nullptr, HSPK, 1536, 1);
    k_gemm_wx<<<384, 256, 0, stream>>>(bwp + 1179648, HSPK, PRE, 384, 1536, 24);
    k_bn_lif<<<1536, 256, 0, stream>>>(PRE, f2bn[i].g, f2bn[i].b, f2bn[i].m, f2bn[i].v, f2b[i], 1.f, FEATS, FEATS, 384, 1);
  }

  // ---- masked pooling + head ----
  k_pool_tokens<<<24, 256, 0, stream>>>(FEATS, MASK, POOLED);
  k_head<<<1, 256, 0, stream>>>(POOLED, hww, hbb, (float*)d_out);
}
